// CustomMultiheadAttention_24172075942051
// MI455X (gfx1250) — matrix operand order repaired and checked by reading, not compiled, not run on hardware
//
#include <hip/hip_runtime.h>

#define HIDDEN   1024
#define NUM_HEADS 16
#define HEAD_DIM  64
#define SEQ      2048
#define BATCH    2

typedef __attribute__((ext_vector_type(16))) __bf16 bf16x16;
typedef __attribute__((ext_vector_type(8)))  __bf16 bf16x8;
typedef __attribute__((ext_vector_type(4)))  __bf16 bf16x4;
typedef __attribute__((ext_vector_type(8)))  float  f32x8;
typedef __attribute__((ext_vector_type(4)))  float  f32x4;

// ---- WMMA fragment builders (wave32 layouts per CDNA5 ISA §7.12.2) ----

// A (16x32 bf16): lane<16 -> row=lane, K {0..7}+{16..23}; lane>=16 -> row=lane-16, K {8..15}+{24..31}
__device__ __forceinline__ bf16x16 fragA_from_bf16(const __bf16* row, int k0, int lane) {
  const int sub = (lane & 16) ? 8 : 0;
  bf16x8 a = *(const bf16x8*)(row + k0 + sub);
  bf16x8 b = *(const bf16x8*)(row + k0 + sub + 16);
  bf16x16 r;
#pragma unroll
  for (int i = 0; i < 8; ++i) { r[i] = a[i]; r[i + 8] = b[i]; }
  return r;
}

__device__ __forceinline__ bf16x16 fragA_from_f32(const float* row, int k0, int lane) {
  const int sub = (lane & 16) ? 8 : 0;
  f32x4 a0 = *(const f32x4*)(row + k0 + sub);
  f32x4 a1 = *(const f32x4*)(row + k0 + sub + 4);
  f32x4 b0 = *(const f32x4*)(row + k0 + sub + 16);
  f32x4 b1 = *(const f32x4*)(row + k0 + sub + 20);
  bf16x16 r;
#pragma unroll
  for (int i = 0; i < 4; ++i) {
    r[i]      = (__bf16)a0[i];  r[i + 4]  = (__bf16)a1[i];
    r[i + 8]  = (__bf16)b0[i];  r[i + 12] = (__bf16)b1[i];
  }
  return r;
}

// B (32x16 bf16): lane<16 -> col=lane, K {0..7}+{16..23}; lane>=16 -> col=lane-16, K {8..15}+{24..31}
__device__ __forceinline__ bf16x16 fragB_from_bf16(const __bf16* col, int k0, int lane) {
  const int off = (lane & 16) ? 8 : 0;
  bf16x8 a = *(const bf16x8*)(col + k0 + off);
  bf16x8 b = *(const bf16x8*)(col + k0 + off + 16);
  bf16x16 r;
#pragma unroll
  for (int i = 0; i < 8; ++i) { r[i] = a[i]; r[i + 8] = b[i]; }
  return r;
}

// ---- Kernel 0: one-time f32 -> bf16 cast (X and W), vectorized b128 load / b64 store ----
__global__ __launch_bounds__(256) void cast_bf16_kernel(
    const float* __restrict__ src, __bf16* __restrict__ dst, int n4)
{
  int i = blockIdx.x * blockDim.x + threadIdx.x;
  const int stride = gridDim.x * blockDim.x;
  for (; i < n4; i += stride) {
    f32x4 v = ((const f32x4*)src)[i];
    bf16x4 o;
#pragma unroll
    for (int j = 0; j < 4; ++j) o[j] = (__bf16)v[j];
    ((bf16x4*)dst)[i] = o;
  }
}

// ---- Kernel A: P = Xbf16 @ Wbf16^T + bias, write in attention-friendly bf16 layout ----
// transposeV=0: dst layout [B, h, S, d]  (q and k)
// transposeV=1: dst layout [B, h, d, S]  (v, so B-fragments of weights@v are K-contiguous)
__global__ __launch_bounds__(128) void qkv_project_kernel(
    const __bf16* __restrict__ X, const __bf16* __restrict__ W,
    const float* __restrict__ bias, __bf16* __restrict__ dst, int transposeV)
{
  const int lane  = threadIdx.x & 31;
  const int wave  = threadIdx.x >> 5;
  const int ntile = blockIdx.x * 4 + wave;   // 0..63  (N = HIDDEN/16)
  const int mtile = blockIdx.y;              // 0..255 (M = B*S/16)
  const int m_lo  = mtile * 16;
  const int nrow  = ntile * 16 + (lane & 15);

  const __bf16* arow = X + (size_t)(m_lo + (lane & 15)) * HIDDEN;  // A row (contiguous K)
  const __bf16* brow = W + (size_t)nrow * HIDDEN;                  // B col = W row (X@W^T)

  f32x8 acc = {};
  for (int k0 = 0; k0 < HIDDEN; k0 += 32) {
    bf16x16 a = fragA_from_bf16(arow, k0, lane);
    bf16x16 b = fragB_from_bf16(brow, k0, lane);
    acc = __builtin_amdgcn_wmma_f32_16x16x32_bf16(false, a, false, b, (short)0, acc, false, false);
  }

  const float bn   = bias[nrow];
  const int head   = nrow >> 6;
  const int dd     = nrow & 63;
  const int mbase  = m_lo + ((lane & 16) ? 8 : 0);  // C: vgpr j -> row j (+8 for hi lanes)
  const int b_idx  = mbase / SEQ;                   // 16-row tile never crosses batch
  const int s0     = mbase % SEQ;

  if (!transposeV) {
    __bf16* p = dst + (((size_t)(b_idx * NUM_HEADS + head) * SEQ + s0) * HEAD_DIM + dd);
#pragma unroll
    for (int j = 0; j < 8; ++j) p[(size_t)j * HEAD_DIM] = (__bf16)(acc[j] + bn);
  } else {
    bf16x8 v;
#pragma unroll
    for (int j = 0; j < 8; ++j) v[j] = (__bf16)(acc[j] + bn);
    *(bf16x8*)(dst + (((size_t)(b_idx * NUM_HEADS + head) * HEAD_DIM + dd) * SEQ + s0)) = v;
  }
}

// ---- Kernel B: scores -> softmax(T)/prune/renorm in LDS -> weights out -> out = W @ V ----
// One block = 4 waves handles a 16-query tile of one (batch, head).
// Dynamic LDS: float sc[16][SEQ] = 128 KB (of 320 KB WGP LDS).
__global__ __launch_bounds__(128) void attention_kernel(
    const __bf16* __restrict__ qbuf, const __bf16* __restrict__ kbuf,
    const __bf16* __restrict__ vbuf, const unsigned char* __restrict__ mask,
    float* __restrict__ out, float* __restrict__ weights)
{
  extern __shared__ float sc[];              // [16][SEQ]
  const int lane  = threadIdx.x & 31;
  const int wave  = threadIdx.x >> 5;
  const int qtile = blockIdx.x;              // 0..127
  const int bh    = blockIdx.y;              // 0..31
  const int batch = bh >> 4;

  // Phase 1: scores tile-by-tile via WMMA, scaled by 1/sqrt(d)/TEMPERATURE = 0.25
  const __bf16* qrow = qbuf + ((size_t)bh * SEQ + qtile * 16 + (lane & 15)) * HEAD_DIM;
  for (int kt = wave; kt < SEQ / 16; kt += 4) {
    const __bf16* krow = kbuf + ((size_t)bh * SEQ + kt * 16 + (lane & 15)) * HEAD_DIM;
    if (kt + 4 < SEQ / 16)
      __builtin_prefetch(krow + 4 * 16 * HEAD_DIM, 0, 0);  // global_prefetch_b8 next K-tile
    f32x8 acc = {};
#pragma unroll
    for (int k0 = 0; k0 < HEAD_DIM; k0 += 32) {
      bf16x16 a = fragA_from_bf16(qrow, k0, lane);
      bf16x16 b = fragB_from_bf16(krow, k0, lane);
      acc = __builtin_amdgcn_wmma_f32_16x16x32_bf16(false, a, false, b, (short)0, acc, false, false);
    }
    const int key = kt * 16 + (lane & 15);
    const bool km = mask[batch * SEQ + key] != 0;
    const int mrow = (lane & 16) ? 8 : 0;
#pragma unroll
    for (int j = 0; j < 8; ++j) {
      float sval = km ? -__builtin_inff() : acc[j] * 0.25f;
      sc[(size_t)(mrow + j) * SEQ + key] = sval;
    }
  }
  __syncthreads();

  // Phase 2: per-row softmax + threshold prune + L1 renorm; 8 lanes per row; all b128 traffic.
  {
    const int row = threadIdx.x >> 3;        // 0..15
    const int seg = threadIdx.x & 7;
    float* r = sc + (size_t)row * SEQ + seg * (SEQ / 8);

    float m = -__builtin_inff();
    for (int i = 0; i < SEQ / 8; i += 4) {
      f32x4 v = *(const f32x4*)(r + i);
      m = fmaxf(m, fmaxf(fmaxf(v[0], v[1]), fmaxf(v[2], v[3])));
    }
    m = fmaxf(m, __shfl_xor(m, 1, 32));
    m = fmaxf(m, __shfl_xor(m, 2, 32));
    m = fmaxf(m, __shfl_xor(m, 4, 32));

    float s = 0.f;
    for (int i = 0; i < SEQ / 8; i += 4) {
      f32x4 v = *(const f32x4*)(r + i);
      f32x4 e;
#pragma unroll
      for (int j = 0; j < 4; ++j) { e[j] = __expf(v[j] - m); s += e[j]; }
      *(f32x4*)(r + i) = e;
    }
    s += __shfl_xor(s, 1, 32); s += __shfl_xor(s, 2, 32); s += __shfl_xor(s, 4, 32);
    const float inv = 1.0f / s;

    float l1 = 0.f;
    for (int i = 0; i < SEQ / 8; i += 4) {
      f32x4 v = *(const f32x4*)(r + i);
      f32x4 w;
#pragma unroll
      for (int j = 0; j < 4; ++j) {
        float x = v[j] * inv;
        x = (x < 0.01f) ? 0.f : x;
        w[j] = x; l1 += x;
      }
      *(f32x4*)(r + i) = w;
    }
    l1 += __shfl_xor(l1, 1, 32); l1 += __shfl_xor(l1, 2, 32); l1 += __shfl_xor(l1, 4, 32);
    const float inv2 = 1.0f / fmaxf(l1, 1e-12f);

    // Final pass: renormalize in LDS and stream weights to HBM with b128 stores.
    float* wout = weights + ((size_t)bh * SEQ + qtile * 16 + row) * SEQ + seg * (SEQ / 8);
    for (int i = 0; i < SEQ / 8; i += 4) {
      f32x4 v = *(const f32x4*)(r + i);
      f32x4 w;
#pragma unroll
      for (int j = 0; j < 4; ++j) w[j] = v[j] * inv2;
      *(f32x4*)(r + i) = w;
      *(f32x4*)(wout + i) = w;
    }
  }
  __syncthreads();

  // Phase 3: out[16 x 64] = weights[16 x SEQ] @ V[SEQ x 64]; wave w owns 16 output channels.
  {
    const int ch = wave * 16 + (lane & 15);
    const __bf16* vrow = vbuf + ((size_t)bh * HEAD_DIM + ch) * SEQ;  // V^T row: K-contiguous
    const float*  arow = sc + (size_t)(lane & 15) * SEQ;
    f32x8 acc = {};
    for (int k0 = 0; k0 < SEQ; k0 += 32) {
      if (k0 + 256 < SEQ) __builtin_prefetch(vrow + k0 + 256, 0, 0);
      bf16x16 a = fragA_from_f32(arow, k0, lane);
      bf16x16 b = fragB_from_bf16(vrow, k0, lane);
      acc = __builtin_amdgcn_wmma_f32_16x16x32_bf16(false, a, false, b, (short)0, acc, false, false);
    }
    const int head  = bh & 15;
    const int mbase = qtile * 16 + ((lane & 16) ? 8 : 0);
    float* p = out + ((size_t)batch * SEQ + mbase) * HIDDEN + head * HEAD_DIM + ch;
#pragma unroll
    for (int j = 0; j < 8; ++j) p[(size_t)j * HIDDEN] = acc[j];
  }
}

extern "C" void kernel_launch(void* const* d_in, const int* in_sizes, int n_in,
                              void* d_out, int out_size, void* d_ws, size_t ws_size,
                              hipStream_t stream) {
  (void)in_sizes; (void)n_in; (void)out_size; (void)ws_size;

  const float* query = (const float*)d_in[0];
  const float* key_  = (const float*)d_in[1];
  const float* value = (const float*)d_in[2];
  const unsigned char* mask = (const unsigned char*)d_in[3];
  const float* Wq = (const float*)d_in[4];
  const float* bq = (const float*)d_in[5];
  const float* Wk = (const float*)d_in[6];
  const float* bk = (const float*)d_in[7];
  const float* Wv = (const float*)d_in[8];
  const float* bv = (const float*)d_in[9];

  float* out     = (float*)d_out;                                  // [B,S,H]
  float* weights = out + (size_t)BATCH * SEQ * HIDDEN;             // [B,h,S,S]

  const size_t nX = (size_t)BATCH * SEQ * HIDDEN;                  // 4,194,304 elems
  const size_t nW = (size_t)HIDDEN * HIDDEN;                       // 1,048,576 elems
  char* ws = (char*)d_ws;
  size_t off = 0;
  __bf16* qbuf  = (__bf16*)(ws + off); off += nX * 2;              // [B,h,S,d]
  __bf16* kbuf  = (__bf16*)(ws + off); off += nX * 2;              // [B,h,S,d]
  __bf16* vbuf  = (__bf16*)(ws + off); off += nX * 2;              // [B,h,d,S]
  __bf16* Xqb   = (__bf16*)(ws + off); off += nX * 2;              // bf16 inputs
  __bf16* Xkb   = (__bf16*)(ws + off); off += nX * 2;
  __bf16* Xvb   = (__bf16*)(ws + off); off += nX * 2;
  __bf16* Wqb   = (__bf16*)(ws + off); off += nW * 2;              // bf16 weights
  __bf16* Wkb   = (__bf16*)(ws + off); off += nW * 2;
  __bf16* Wvb   = (__bf16*)(ws + off); off += nW * 2;

  // One-time casts (b128-load / b64-store streams)
  dim3 bC(256);
  dim3 gCX((unsigned)((nX / 4 + 255) / 256 < 2048 ? (nX / 4 + 255) / 256 : 2048));
  dim3 gCW((unsigned)((nW / 4 + 255) / 256 < 2048 ? (nW / 4 + 255) / 256 : 2048));
  hipLaunchKernelGGL(cast_bf16_kernel, gCX, bC, 0, stream, query, Xqb, (int)(nX / 4));
  hipLaunchKernelGGL(cast_bf16_kernel, gCX, bC, 0, stream, key_,  Xkb, (int)(nX / 4));
  hipLaunchKernelGGL(cast_bf16_kernel, gCX, bC, 0, stream, value, Xvb, (int)(nX / 4));
  hipLaunchKernelGGL(cast_bf16_kernel, gCW, bC, 0, stream, Wq, Wqb, (int)(nW / 4));
  hipLaunchKernelGGL(cast_bf16_kernel, gCW, bC, 0, stream, Wk, Wkb, (int)(nW / 4));
  hipLaunchKernelGGL(cast_bf16_kernel, gCW, bC, 0, stream, Wv, Wvb, (int)(nW / 4));

  // Projections (pure bf16 WMMA GEMMs)
  dim3 gA(16, 256), bA(128);
  hipLaunchKernelGGL(qkv_project_kernel, gA, bA, 0, stream, Xqb, Wqb, bq, qbuf, 0);
  hipLaunchKernelGGL(qkv_project_kernel, gA, bA, 0, stream, Xkb, Wkb, bk, kbuf, 0);
  hipLaunchKernelGGL(qkv_project_kernel, gA, bA, 0, stream, Xvb, Wvb, bv, vbuf, 1);

  // Attention with LDS-resident score block
  dim3 gB(SEQ / 16, BATCH * NUM_HEADS), bB(128);
  hipLaunchKernelGGL(attention_kernel, gB, bB, 16 * SEQ * sizeof(float), stream,
                     qbuf, kbuf, vbuf, mask, out, weights);
}
